// EncoderCellRNN_52501680226810
// MI455X (gfx1250) — compile-verified, matrix operand order checked by reading, not run on hardware
//
#include <hip/hip_runtime.h>

// ---------------------------------------------------------------------------
// GRU encoder for MI455X (gfx1250): bf16 WMMA (v_wmma_f32_16x16x32_bf16),
// f32 accumulate + f32 gate math. wave32. Software-pipelined fragment loads,
// pinned with tied-operand asm barriers so WMMAs cannot hoist above the
// next-iteration loads (and loads cannot sink below them).
// ---------------------------------------------------------------------------

#define BATCH   64
#define SRCLEN  512
#define EMBED   1024
#define HIDDEN  1024
#define THREEH  3072   // 3*HIDDEN

typedef __attribute__((ext_vector_type(16))) __bf16 v16bf;
typedef __attribute__((ext_vector_type(8)))  __bf16 v8bf;
typedef __attribute__((ext_vector_type(4)))  __bf16 v4bf;
typedef __attribute__((ext_vector_type(8)))  float  v8f;
typedef __attribute__((ext_vector_type(8)))  int    v8i;

// Per CDNA5 ISA 16-bit A-layout (16x32): lane m (<16) holds K {0..7,16..23},
// lane m+16 holds K {8..15,24..31} of row (lane&15). With koff=(lane>>4)*8
// the fragment is exactly two 16B loads at [k0+koff] and [k0+koff+16].
__device__ __forceinline__ v16bf ldfrag(const __bf16* p) {
    v8bf lo = *(const v8bf*)p;
    v8bf hi = *(const v8bf*)(p + 16);
    v16bf r;
#pragma unroll
    for (int i = 0; i < 8; ++i) { r[i] = lo[i]; r[i + 8] = hi[i]; }
    return r;
}

// Pipeline barrier: current fragments flow THROUGH the asm (tied "+v"), so
// the consuming WMMAs cannot be scheduled above it; the "memory" clobber
// keeps the already-issued next-iteration loads above it. Together this
// forces a genuine 1-deep software pipeline with two live fragment sets.
__device__ __forceinline__ void pin_frags4(v16bf& a, v16bf& b0, v16bf& b1,
                                           v16bf& b2, v16bf& b3) {
    v8i ta = __builtin_bit_cast(v8i, a);
    v8i t0 = __builtin_bit_cast(v8i, b0);
    v8i t1 = __builtin_bit_cast(v8i, b1);
    v8i t2 = __builtin_bit_cast(v8i, b2);
    v8i t3 = __builtin_bit_cast(v8i, b3);
    asm volatile("" : "+v"(ta), "+v"(t0), "+v"(t1), "+v"(t2), "+v"(t3) :: "memory");
    a  = __builtin_bit_cast(v16bf, ta);
    b0 = __builtin_bit_cast(v16bf, t0);
    b1 = __builtin_bit_cast(v16bf, t1);
    b2 = __builtin_bit_cast(v16bf, t2);
    b3 = __builtin_bit_cast(v16bf, t3);
}

__device__ __forceinline__ void pin_frags3(v16bf& a, v16bf& b0, v16bf& b1,
                                           v16bf& b2) {
    v8i ta = __builtin_bit_cast(v8i, a);
    v8i t0 = __builtin_bit_cast(v8i, b0);
    v8i t1 = __builtin_bit_cast(v8i, b1);
    v8i t2 = __builtin_bit_cast(v8i, b2);
    asm volatile("" : "+v"(ta), "+v"(t0), "+v"(t1), "+v"(t2) :: "memory");
    a  = __builtin_bit_cast(v16bf, ta);
    b0 = __builtin_bit_cast(v16bf, t0);
    b1 = __builtin_bit_cast(v16bf, t1);
    b2 = __builtin_bit_cast(v16bf, t2);
}

__device__ __forceinline__ float sigmoidf_(float x) {
    return 1.0f / (1.0f + __expf(-x));
}

#define WMMA_BF16(A, B, C) \
    __builtin_amdgcn_wmma_f32_16x16x32_bf16(false, (A), false, (B), (short)0, (C), false, false)

// ---------------------------------------------------------------------------
// 1) f32 -> bf16 weight conversion (vectorized)
// ---------------------------------------------------------------------------
__global__ void k_cvt_bf16(const float* __restrict__ src,
                           __bf16* __restrict__ dst, int n) {
    int i = (blockIdx.x * 256 + threadIdx.x) * 4;
    if (i < n) {
        float4 v = *(const float4*)(src + i);
        v4bf o;
        o[0] = (__bf16)v.x; o[1] = (__bf16)v.y;
        o[2] = (__bf16)v.z; o[3] = (__bf16)v.w;
        *(v4bf*)(dst + i) = o;
    }
}

// ---------------------------------------------------------------------------
// 2) Embedding gather + bf16 convert, time-major X[s*B + b][E]
// ---------------------------------------------------------------------------
__global__ void k_embed(const int* __restrict__ src,
                        const float* __restrict__ Wemb,
                        __bf16* __restrict__ X) {
    int bs = blockIdx.x;          // = s*BATCH + b
    int s  = bs >> 6;             // / BATCH
    int b  = bs & 63;
    int tok = src[b * SRCLEN + s];
    int e   = threadIdx.x * 4;
    float4 v = *(const float4*)(Wemb + (size_t)tok * EMBED + e);
    v4bf o;
    o[0] = (__bf16)v.x; o[1] = (__bf16)v.y;
    o[2] = (__bf16)v.z; o[3] = (__bf16)v.w;
    *(v4bf*)(X + (size_t)bs * EMBED + e) = o;
}

// ---------------------------------------------------------------------------
// 3) Big parallel GEMM: Gi[row][n] = sum_k X[row][k]*Wih[n][k] + b_ih[n]
//    M=32768 (time-major rows), N=3072, K=1024.
//    One wave computes a 16(M) x 64(N) tile: 4 f32 accumulators, A reused.
// ---------------------------------------------------------------------------
__global__ void __launch_bounds__(256)
k_gemm_gi(const __bf16* __restrict__ X, const __bf16* __restrict__ W,
          const float* __restrict__ bias, void* __restrict__ Gi, int gi_f32) {
    const int lane  = threadIdx.x & 31;
    const int wv    = threadIdx.x >> 5;
    const int mt    = blockIdx.x;                 // 0..2047  (M tile)
    const int ng    = blockIdx.y * 8 + wv;        // 0..47    (N group of 64)
    const int nbase = ng * 64;
    const int koff  = (lane >> 4) * 8;
    const int l15   = lane & 15;

    const __bf16* arow = X + (size_t)(mt * 16 + l15) * EMBED + koff;
    const __bf16* br0  = W + (size_t)(nbase +  0 + l15) * EMBED + koff;
    const __bf16* br1  = W + (size_t)(nbase + 16 + l15) * EMBED + koff;
    const __bf16* br2  = W + (size_t)(nbase + 32 + l15) * EMBED + koff;
    const __bf16* br3  = W + (size_t)(nbase + 48 + l15) * EMBED + koff;

    v8f acc[4];
#pragma unroll
    for (int i = 0; i < 4; ++i) acc[i] = (v8f){0.f,0.f,0.f,0.f,0.f,0.f,0.f,0.f};

    // --- software pipeline: prologue loads for k0 = 0 ---
    v16bf aC  = ldfrag(arow);
    v16bf bC0 = ldfrag(br0);
    v16bf bC1 = ldfrag(br1);
    v16bf bC2 = ldfrag(br2);
    v16bf bC3 = ldfrag(br3);

#pragma unroll 2
    for (int k0 = 32; k0 < EMBED; k0 += 32) {
        // issue next iteration's loads BEFORE consuming current fragments
        v16bf aN  = ldfrag(arow + k0);
        v16bf bN0 = ldfrag(br0 + k0);
        v16bf bN1 = ldfrag(br1 + k0);
        v16bf bN2 = ldfrag(br2 + k0);
        v16bf bN3 = ldfrag(br3 + k0);
        pin_frags4(aC, bC0, bC1, bC2, bC3);  // WMMAs below, loads above
        acc[0] = WMMA_BF16(aC, bC0, acc[0]);
        acc[1] = WMMA_BF16(aC, bC1, acc[1]);
        acc[2] = WMMA_BF16(aC, bC2, acc[2]);
        acc[3] = WMMA_BF16(aC, bC3, acc[3]);
        aC = aN; bC0 = bN0; bC1 = bN1; bC2 = bN2; bC3 = bN3;
    }
    acc[0] = WMMA_BF16(aC, bC0, acc[0]);
    acc[1] = WMMA_BF16(aC, bC1, acc[1]);
    acc[2] = WMMA_BF16(aC, bC2, acc[2]);
    acc[3] = WMMA_BF16(aC, bC3, acc[3]);

    // C/D layout: VGPR r -> row (r + 8*(lane>>4)), col (lane&15) within tile.
    const int rbase = (lane >> 4) * 8;
#pragma unroll
    for (int i = 0; i < 4; ++i) {
        const int col = nbase + i * 16 + l15;
        const float bb = bias[col];
#pragma unroll
        for (int r = 0; r < 8; ++r) {
            const size_t idx = (size_t)(mt * 16 + rbase + r) * THREEH + col;
            const float v = acc[i][r] + bb;
            if (gi_f32) ((float*)Gi)[idx] = v;
            else        ((__bf16*)Gi)[idx] = (__bf16)v;
        }
    }
}

// ---------------------------------------------------------------------------
// 4) zero the bf16 h buffer (t=0 state)
// ---------------------------------------------------------------------------
__global__ void k_zero32(unsigned int* __restrict__ p, int nwords) {
    int i = blockIdx.x * 256 + threadIdx.x;
    if (i < nwords) p[i] = 0u;
}

// ---------------------------------------------------------------------------
// 5) One GRU step. wave = 16(batch) x 16(j) output tile across all 3 gates:
//    3 pipelined WMMA chains over K=1024 (W_hh hot in L2: 6 MB bf16).
//    grid = (4, 8), block = 256 (8 waves): mt = blockIdx.x, jt = by*8 + wv.
// ---------------------------------------------------------------------------
__global__ void __launch_bounds__(256)
k_gru_step(const __bf16* __restrict__ hprev_bf, const __bf16* __restrict__ Whh,
           const float* __restrict__ bhh, const void* __restrict__ Gi, int gi_f32,
           float* __restrict__ outputs, __bf16* __restrict__ hnext_bf, int t) {
    const int lane  = threadIdx.x & 31;
    const int wv    = threadIdx.x >> 5;
    const int mt    = blockIdx.x;               // 0..3  (batch tile)
    const int jt    = blockIdx.y * 8 + wv;      // 0..63 (hidden tile)
    const int koff  = (lane >> 4) * 8;
    const int l15   = lane & 15;
    const int jlane = jt * 16 + l15;            // 0..1023

    const __bf16* arow = hprev_bf + (size_t)(mt * 16 + l15) * HIDDEN + koff;
    const __bf16* br   = Whh + (size_t)(           jlane) * HIDDEN + koff;
    const __bf16* bz   = Whh + (size_t)(HIDDEN   + jlane) * HIDDEN + koff;
    const __bf16* bn   = Whh + (size_t)(2*HIDDEN + jlane) * HIDDEN + koff;

    v8f aR = (v8f){0.f,0.f,0.f,0.f,0.f,0.f,0.f,0.f};
    v8f aZ = aR, aN = aR;

    // --- software pipeline ---
    v16bf aC  = ldfrag(arow);
    v16bf fCr = ldfrag(br);
    v16bf fCz = ldfrag(bz);
    v16bf fCn = ldfrag(bn);

#pragma unroll 2
    for (int k0 = 32; k0 < HIDDEN; k0 += 32) {
        v16bf aNx = ldfrag(arow + k0);
        v16bf fNr = ldfrag(br + k0);
        v16bf fNz = ldfrag(bz + k0);
        v16bf fNn = ldfrag(bn + k0);
        pin_frags3(aC, fCr, fCz, fCn);       // WMMAs below, loads above
        aR = WMMA_BF16(aC, fCr, aR);
        aZ = WMMA_BF16(aC, fCz, aZ);
        aN = WMMA_BF16(aC, fCn, aN);
        aC = aNx; fCr = fNr; fCz = fNz; fCn = fNn;
    }
    aR = WMMA_BF16(aC, fCr, aR);
    aZ = WMMA_BF16(aC, fCz, aZ);
    aN = WMMA_BF16(aC, fCn, aN);

    const float bhr = bhh[jlane];
    const float bhz = bhh[HIDDEN + jlane];
    const float bhn = bhh[2*HIDDEN + jlane];
    const int rbase = (lane >> 4) * 8;

#pragma unroll
    for (int r = 0; r < 8; ++r) {
        const int b = mt * 16 + rbase + r;                     // batch index
        const size_t girow = ((size_t)t * BATCH + b) * THREEH; // time-major Gi
        float gir, giz, gin;
        if (gi_f32) {
            const float* g = (const float*)Gi + girow;
            gir = g[jlane]; giz = g[HIDDEN + jlane]; gin = g[2*HIDDEN + jlane];
        } else {
            const __bf16* g = (const __bf16*)Gi + girow;
            gir = (float)g[jlane]; giz = (float)g[HIDDEN + jlane]; gin = (float)g[2*HIDDEN + jlane];
        }
        const float rg = sigmoidf_(gir + aR[r] + bhr);
        const float zg = sigmoidf_(giz + aZ[r] + bhz);
        const float ng = tanhf(gin + rg * (aN[r] + bhn));
        const size_t obase = (size_t)b * SRCLEN * HIDDEN;
        const float hp = (t == 0) ? 0.0f
                                  : outputs[obase + (size_t)(t - 1) * HIDDEN + jlane];
        const float hn = (1.0f - zg) * ng + zg * hp;
        outputs[obase + (size_t)t * HIDDEN + jlane] = hn;      // canonical f32 h
        hnext_bf[(size_t)b * HIDDEN + jlane] = (__bf16)hn;     // GEMM operand
    }
}

// ---------------------------------------------------------------------------
// 6) hT = outputs[:, SRCLEN-1, :]
// ---------------------------------------------------------------------------
__global__ void k_copy_hT(const float* __restrict__ outputs,
                          float* __restrict__ hT) {
    int idx = blockIdx.x * 256 + threadIdx.x;     // 0..65535
    int b = idx >> 10, j = idx & 1023;
    hT[idx] = outputs[(size_t)b * SRCLEN * HIDDEN + (size_t)(SRCLEN - 1) * HIDDEN + j];
}

// ---------------------------------------------------------------------------
extern "C" void kernel_launch(void* const* d_in, const int* in_sizes, int n_in,
                              void* d_out, int out_size, void* d_ws, size_t ws_size,
                              hipStream_t stream) {
    (void)in_sizes; (void)n_in; (void)out_size;
    const int*   src  = (const int*)  d_in[0];
    const float* Wemb = (const float*)d_in[1];
    const float* Wih  = (const float*)d_in[2];
    const float* Whh  = (const float*)d_in[3];
    const float* bih  = (const float*)d_in[4];
    const float* bhh  = (const float*)d_in[5];

    float* outputs = (float*)d_out;
    float* hT      = outputs + (size_t)BATCH * SRCLEN * HIDDEN;

    // Workspace carve-up (256B aligned).
    char* ws = (char*)d_ws;
    size_t off = 0;
    auto take = [&](size_t bytes) {
        size_t r = off; off = (off + bytes + 255) & ~(size_t)255; return r;
    };
    __bf16* Xbf   = (__bf16*)(ws + take((size_t)BATCH * SRCLEN * EMBED * 2));
    __bf16* WihBf = (__bf16*)(ws + take((size_t)THREEH * EMBED * 2));
    __bf16* WhhBf = (__bf16*)(ws + take((size_t)THREEH * HIDDEN * 2));
    __bf16* h0    = (__bf16*)(ws + take((size_t)BATCH * HIDDEN * 2));
    __bf16* h1    = (__bf16*)(ws + take((size_t)BATCH * HIDDEN * 2));
    const size_t gi_elems = (size_t)BATCH * SRCLEN * THREEH;
    void* Gi = (void*)(ws + off);
    const int gi_f32 = (off + gi_elems * 4 <= ws_size) ? 1 : 0; // f32 if it fits, else bf16

    // Phase 1: precision conversion + embedding gather (fully parallel).
    k_cvt_bf16<<<(THREEH * EMBED) / 1024, 256, 0, stream>>>(Wih, WihBf, THREEH * EMBED);
    k_cvt_bf16<<<(THREEH * HIDDEN) / 1024, 256, 0, stream>>>(Whh, WhhBf, THREEH * HIDDEN);
    k_embed<<<BATCH * SRCLEN, 256, 0, stream>>>(src, Wemb, Xbf);

    // Phase 2: the big input-side GEMM, Gi = X * Wih^T + b_ih (WMMA bf16).
    k_gemm_gi<<<dim3((BATCH * SRCLEN) / 16, 6), 256, 0, stream>>>(Xbf, WihBf, bih, Gi, gi_f32);

    // Phase 3: sequential recurrence, 512 dependent steps (WMMA bf16).
    k_zero32<<<(BATCH * HIDDEN * 2 / 4 + 255) / 256, 256, 0, stream>>>((unsigned int*)h0,
                                                                      BATCH * HIDDEN * 2 / 4);
    for (int t = 0; t < SRCLEN; ++t) {
        __bf16* hp = (t & 1) ? h1 : h0;
        __bf16* hn = (t & 1) ? h0 : h1;
        k_gru_step<<<dim3(4, 8), 256, 0, stream>>>(hp, WhhBf, bhh, Gi, gi_f32,
                                                   outputs, hn, t);
    }

    // Phase 4: final hidden state.
    k_copy_hT<<<(BATCH * HIDDEN) / 256, 256, 0, stream>>>(outputs, hT);
}
